// EdgeDecoder_30176440221878
// MI455X (gfx1250) — matrix operand order repaired and checked by reading, not compiled, not run on hardware
//
#include <hip/hip_runtime.h>
#include <hip/hip_bf16.h>

// ---------------------------------------------------------------------------
// EdgeDecoder for MI455X (gfx1250, wave32, WMMA + pipelined async LDS DMA)
//   out[e] = W2 . relu(W1 @ concat(z[u[e]], z[v[e]]) + b1) + b2
// f16 A/B, f32 accum via v_wmma_f32_16x16x32_f16.
//  - z -> f16 once in d_ws (25.6MB, L2-resident); edge gather becomes
//    GLOBAL_LOAD_ASYNC_TO_LDS_B128 (ASYNCcnt), double-buffered across chunks
//    so the DMA for chunk c+1 flies under the WMMA phase of chunk c
//    (in-order async completion => s_wait_asynccnt 0x4 releases the current
//    buffer while the prefetch stays outstanding).
//  - W1 -> f16 once in d_ws; each wave pins its K=256 B-slice in 64 VGPRs.
//  - Epilogue: multi-value butterfly (8 bpermutes/M-tile) + 1 ds_add_f32.
//  - Fallback path (ws too small for z_f16): convert-in-register staging.
// ---------------------------------------------------------------------------

#define NNODES  100000
#define EMB     128
#define KDIM    256        // 2*EMB
#define HID     256
#define NEDGE   500000
#define MBLK    64         // edges per chunk per block
#define TPB     512        // 16 waves; wave w owns hidden units [16w, 16w+16)
#define NCHUNK  ((NEDGE + MBLK - 1) / MBLK)   // 7813
#define GRID    1024
#define ASTR    264        // LDS row stride in halves (528 B: 16B-aligned, padded)

#define W1H_BYTES   (HID * KDIM * 2)                 // 131072
#define ZH_BYTES    ((size_t)NNODES * EMB * 2)       // 25,600,000
#define WS_NEEDED   (W1H_BYTES + ZH_BYTES)

typedef __attribute__((ext_vector_type(16))) _Float16     v16h;
typedef __attribute__((ext_vector_type(8)))  float        v8f;
typedef __attribute__((ext_vector_type(4)))  unsigned int v4u;
typedef __attribute__((ext_vector_type(4)))  _Float16     v4h;
typedef __attribute__((ext_vector_type(4)))  float        v4f;

union HFrag { v16h h; v4u u[2]; };

// Butterfly merge: combine two independent partial sums while reducing across
// the xor-`mask` lane pair.  Lanes with (lane&mask)==0 end up carrying the
// pair-sum of `a`; lanes with the bit set carry the pair-sum of `b`.
__device__ __forceinline__ float merge_pair(float a, float b, int mask, int lane) {
    const bool hi = (lane & mask) != 0;
    float send = hi ? a : b;
    float keep = hi ? b : a;
    float got  = __shfl_xor(send, mask, 32);
    return keep + got;
}

// --- prep: fp32 -> f16 converters (outputs live in d_ws, L2-resident) -------
__global__ void cvt_w1_f16(const float* __restrict__ W1, _Float16* __restrict__ W1h) {
    int i = blockIdx.x * blockDim.x + threadIdx.x;   // 65536 total
    W1h[i] = (_Float16)W1[i];
}

__global__ void cvt_z_f16(const float* __restrict__ z, _Float16* __restrict__ zh) {
    int i = blockIdx.x * blockDim.x + threadIdx.x;   // 3,200,000 x 4 elements
    v4f f = ((const v4f*)z)[i];
    v4h h;
    h.x = (_Float16)f.x; h.y = (_Float16)f.y;
    h.z = (_Float16)f.z; h.w = (_Float16)f.w;
    ((v4h*)zh)[i] = h;
}

// --- staging helpers (per-thread: 64 B span of one edge row) ----------------
__device__ __forceinline__ void stage_async_tile(const _Float16* __restrict__ zh,
                                                 const long long* __restrict__ eidx,
                                                 int base, int srow, int sseg,
                                                 _Float16* dst) {
    long long e = (long long)base + srow;
    if (e >= NEDGE) e = NEDGE - 1;              // clamp: keeps EXEC full
    const long long uu = eidx[e];
    const long long vv = eidx[NEDGE + e];
    const _Float16* src = (sseg < 4) ? (zh + uu * EMB + sseg * 32)
                                     : (zh + vv * EMB + (sseg - 4) * 32);
    unsigned ldsOff = (unsigned)(size_t)dst;    // LDS byte offset (addr[31:0])
    // `offset:` applies to BOTH the LDS and the global address (ISA 08 §4.4)
    asm volatile("global_load_async_to_lds_b128 %0, %1, off"
                 :: "v"(ldsOff), "v"(src) : "memory");
    asm volatile("global_load_async_to_lds_b128 %0, %1, off offset:16"
                 :: "v"(ldsOff), "v"(src) : "memory");
    asm volatile("global_load_async_to_lds_b128 %0, %1, off offset:32"
                 :: "v"(ldsOff), "v"(src) : "memory");
    asm volatile("global_load_async_to_lds_b128 %0, %1, off offset:48"
                 :: "v"(ldsOff), "v"(src) : "memory");
}

__device__ __forceinline__ void stage_cvt_tile(const float* __restrict__ z,
                                               const long long* __restrict__ eidx,
                                               int base, int srow, int sseg,
                                               _Float16* dst) {
    long long e = (long long)base + srow;
    if (e >= NEDGE) e = NEDGE - 1;
    const long long uu = eidx[e];
    const long long vv = eidx[NEDGE + e];
    const float* src = (sseg < 4) ? (z + uu * EMB + sseg * 32)
                                  : (z + vv * EMB + (sseg - 4) * 32);
#pragma unroll
    for (int i = 0; i < 8; ++i) {
        v4f f = *(const v4f*)(src + i * 4);
        v4h h;
        h.x = (_Float16)f.x; h.y = (_Float16)f.y;
        h.z = (_Float16)f.z; h.w = (_Float16)f.w;
        *(v4h*)(dst + i * 4) = h;
    }
}

// --- per-chunk compute: 4 M-tiles x (8 x wmma 16x16x32) + fused epilogue ----
__device__ __forceinline__ void compute_tile(const _Float16* __restrict__ smBase,
                                             const HFrag (&B)[8],
                                             int am, int akb, int lane,
                                             float b1v, float w2v,
                                             float* edgeAcc) {
#pragma unroll
    for (int mt = 0; mt < 4; ++mt) {
        v8f acc = {0.f, 0.f, 0.f, 0.f, 0.f, 0.f, 0.f, 0.f};
        const _Float16* arow = smBase + (mt * 16 + am) * ASTR;

        // double-buffered A-fragment loads (overlap ds latency with WMMA)
        HFrag A0, A1;
        A0.u[0] = *(const v4u*)(arow + akb);
        A0.u[1] = *(const v4u*)(arow + akb + 16);
#pragma unroll
        for (int ks = 0; ks < 8; ++ks) {
            if (ks < 7) {
                const _Float16* ap = arow + (ks + 1) * 32 + akb;
                A1.u[0] = *(const v4u*)(ap);
                A1.u[1] = *(const v4u*)(ap + 16);
            }
            acc = __builtin_amdgcn_wmma_f32_16x16x32_f16(
                false, A0.h, false, B[ks].h, (short)0, acc, false, false);
            A0 = A1;
        }

        // bias + relu + dot W2, butterfly-reduce the 16-lane halves
        // C/D layout: VGPR r -> row (r | r+8 by lane half), lane&15 -> col n
        float v0 = fmaxf(acc[0] + b1v, 0.f) * w2v;
        float v1 = fmaxf(acc[1] + b1v, 0.f) * w2v;
        float v2 = fmaxf(acc[2] + b1v, 0.f) * w2v;
        float v3 = fmaxf(acc[3] + b1v, 0.f) * w2v;
        float v4 = fmaxf(acc[4] + b1v, 0.f) * w2v;
        float v5 = fmaxf(acc[5] + b1v, 0.f) * w2v;
        float v6 = fmaxf(acc[6] + b1v, 0.f) * w2v;
        float v7 = fmaxf(acc[7] + b1v, 0.f) * w2v;
        float s0 = merge_pair(v0, v1, 1, lane);
        float s1 = merge_pair(v2, v3, 1, lane);
        float s2 = merge_pair(v4, v5, 1, lane);
        float s3 = merge_pair(v6, v7, 1, lane);
        float t0 = merge_pair(s0, s1, 2, lane);
        float t1 = merge_pair(s2, s3, 2, lane);
        float q  = merge_pair(t0, t1, 4, lane);
        q += __shfl_xor(q, 8, 32);
        if ((lane & 8) == 0) {                   // lanes 0-7 / 16-23: rows
            int m = mt * 16 + (lane & 7) + ((lane >= 16) ? 8 : 0);
            atomicAdd(&edgeAcc[m], q);           // ds_add_f32 across waves
        }
    }
}

// --- main kernel ------------------------------------------------------------
template <bool ASYNC>
__global__ __launch_bounds__(TPB)
void edge_decoder_wmma(const float* __restrict__ z,        // fallback source
                       const _Float16* __restrict__ zh,    // async source
                       const long long* __restrict__ eidx, // [2, NEDGE] int64
                       const float* __restrict__ b1,
                       const float* __restrict__ W2,
                       const float* __restrict__ b2,
                       const _Float16* __restrict__ W1h,
                       float* __restrict__ out) {
    __shared__ _Float16 smA[2][MBLK * ASTR];   // 2 x 33792 B ping-pong tiles
    __shared__ float    edgeAcc[MBLK];

    const int t    = threadIdx.x;
    const int lane = t & 31;
    const int wave = t >> 5;               // 0..15 -> hidden tile

    // --- per-lane constants for this wave's N-tile ---
    const int n     = lane & 15;
    const int nglob = wave * 16 + n;       // hidden unit handled by this lane
    const float b1v = b1[nglob];
    const float w2v = W2[nglob];
    const float b2v = b2[0];

    // --- preload ALL B fragments for this wave's N-tile (held in VGPRs) -----
    // 16-bit B (32x16) layout: lanes 0-15 hold K=0..7 / 16..23 of the kstep,
    // lanes 16-31 hold K=8..15 / 24..31.  Row n of B = W1[nglob][k].
    const int kbB = (lane >= 16) ? 8 : 0;
    HFrag B[8];
#pragma unroll
    for (int ks = 0; ks < 8; ++ks) {
        const _Float16* p = W1h + nglob * KDIM + ks * 32 + kbB;
        B[ks].u[0] = *(const v4u*)(p);         // halves kb .. kb+7
        B[ks].u[1] = *(const v4u*)(p + 16);    // halves kb+16 .. kb+23
    }

    // A-frag addressing (16-bit A 16x32 layout):
    //   lanes 0-15: m=lane, K = K0..K0+7 / K0+16..K0+23, K0=32*ks
    //   lanes 16-31: same m, K0 += 8
    const int am  = lane & 15;
    const int akb = (lane >= 16) ? 8 : 0;

    // staging geometry (per thread): row = edge-in-chunk, seg = 32-col span
    const int srow = t >> 3;               // 0..63
    const int sseg = t & 7;                // 0..7
    const int soff = srow * ASTR + sseg * 32;

    if (ASYNC) {
        // ---- software-pipelined: DMA chunk c+stride under WMMA of chunk c ----
        int c = blockIdx.x;
        int p = 0;
        if (c < NCHUNK)
            stage_async_tile(zh, eidx, c * MBLK, srow, sseg, &smA[0][soff]);
        for (; c < NCHUNK; c += gridDim.x, p ^= 1) {
            const int cn = c + (int)gridDim.x;
            if (cn < NCHUNK) {
                stage_async_tile(zh, eidx, cn * MBLK, srow, sseg, &smA[p ^ 1][soff]);
                // async loads complete in order: <=4 outstanding means the
                // current buffer's 4 DMAs retired; the prefetch stays in flight
                asm volatile("s_wait_asynccnt 0x4" ::: "memory");
            } else {
                asm volatile("s_wait_asynccnt 0x0" ::: "memory");
            }
            if (t < MBLK) edgeAcc[t] = b2v;             // seed with bias2
            __syncthreads();

            compute_tile(&smA[p][0], B, am, akb, lane, b1v, w2v, edgeAcc);
            __syncthreads();

            if (t < MBLK) {
                int e = c * MBLK + t;
                if (e < NEDGE) out[e] = edgeAcc[t];
            }
            __syncthreads();   // close out reads of smA[p^1]/edgeAcc
        }
    } else {
        // ---- fallback: single-buffer, convert-in-register staging ----
        for (int c = blockIdx.x; c < NCHUNK; c += gridDim.x) {
            stage_cvt_tile(z, eidx, c * MBLK, srow, sseg, &smA[0][soff]);
            if (t < MBLK) edgeAcc[t] = b2v;
            __syncthreads();

            compute_tile(&smA[0][0], B, am, akb, lane, b1v, w2v, edgeAcc);
            __syncthreads();

            if (t < MBLK) {
                int e = c * MBLK + t;
                if (e < NEDGE) out[e] = edgeAcc[t];
            }
            __syncthreads();
        }
    }
}

// ---------------------------------------------------------------------------
extern "C" void kernel_launch(void* const* d_in, const int* in_sizes, int n_in,
                              void* d_out, int out_size, void* d_ws, size_t ws_size,
                              hipStream_t stream) {
    const float*     zp   = (const float*)d_in[0];
    const long long* eidx = (const long long*)d_in[1];
    const float*     W1   = (const float*)d_in[2];
    const float*     b1   = (const float*)d_in[3];
    const float*     W2   = (const float*)d_in[4];
    const float*     b2   = (const float*)d_in[5];
    float*           out  = (float*)d_out;
    _Float16*        W1h  = (_Float16*)d_ws;                      // 128 KB
    _Float16*        zh   = (_Float16*)((char*)d_ws + W1H_BYTES); // 25.6 MB

    hipLaunchKernelGGL(cvt_w1_f16, dim3((HID * KDIM) / 256), dim3(256), 0, stream,
                       W1, W1h);

    if (ws_size >= WS_NEEDED) {
        hipLaunchKernelGGL(cvt_z_f16, dim3((NNODES * EMB) / (4 * 256)), dim3(256),
                           0, stream, zp, zh);
        hipLaunchKernelGGL(edge_decoder_wmma<true>, dim3(GRID), dim3(TPB), 0, stream,
                           zp, zh, eidx, b1, W2, b2, W1h, out);
    } else {
        hipLaunchKernelGGL(edge_decoder_wmma<false>, dim3(GRID), dim3(TPB), 0, stream,
                           zp, zh, eidx, b1, W2, b2, W1h, out);
    }
}
